// Network_64201171140960
// MI455X (gfx1250) — matrix operand order repaired and checked by reading, not compiled, not run on hardware
//
#include <hip/hip_runtime.h>
#include <math.h>

#define BATCH 8
#define NPTS  4096

typedef _Float16 f16;
typedef __attribute__((ext_vector_type(16))) _Float16 v16h;
typedef __attribute__((ext_vector_type(8)))  _Float16 v8h;
typedef __attribute__((ext_vector_type(8)))  float    v8f;

// ---------------------------------------------------------------------------
// Preprocess: pcs(B,N,4) -> xyz(B,N,3), feats(B,N,4)=[x,y,z,w]; point0 = (cp,1)
// ---------------------------------------------------------------------------
__global__ void k_pre(const float* __restrict__ pcs, const float* __restrict__ cp,
                      float* __restrict__ xyz, float* __restrict__ feat) {
  int t = blockIdx.x * blockDim.x + threadIdx.x;
  if (t >= BATCH * NPTS) return;
  int b = t / NPTS, n = t % NPTS;
  float x, y, z, w;
  if (n == 0) {
    x = cp[b * 3 + 0]; y = cp[b * 3 + 1]; z = cp[b * 3 + 2]; w = 1.f;
  } else {
    const float* p = pcs + (size_t)t * 4;
    x = p[0]; y = p[1]; z = p[2]; w = p[3];
  }
  xyz[t * 3 + 0] = x; xyz[t * 3 + 1] = y; xyz[t * 3 + 2] = z;
  feat[t * 4 + 0] = x; feat[t * 4 + 1] = y; feat[t * 4 + 2] = z; feat[t * 4 + 3] = w;
}

// ---------------------------------------------------------------------------
// Farthest point sampling: one workgroup (256 thr) per batch, dists in LDS.
// ---------------------------------------------------------------------------
__global__ void k_fps(const float* __restrict__ xyz, int* __restrict__ idx,
                      int Ns, int np) {
  __shared__ float dists[4096];
  __shared__ float smax[256];
  __shared__ int   simax[256];
  __shared__ int   sfar;
  int b = blockIdx.x;
  const float* X = xyz + (size_t)b * Ns * 3;
  for (int j = threadIdx.x; j < Ns; j += blockDim.x) dists[j] = 1e10f;
  if (threadIdx.x == 0) sfar = 0;
  __syncthreads();
  for (int it = 0; it < np; ++it) {
    int far = sfar;
    if (threadIdx.x == 0) idx[b * np + it] = far;
    float cx = X[far * 3 + 0], cy = X[far * 3 + 1], cz = X[far * 3 + 2];
    float best = -1.f; int bi = 0;
    for (int j = threadIdx.x; j < Ns; j += blockDim.x) {
      float dx = X[j * 3 + 0] - cx, dy = X[j * 3 + 1] - cy, dz = X[j * 3 + 2] - cz;
      float d = dx * dx + dy * dy + dz * dz;
      float m = fminf(dists[j], d);
      dists[j] = m;
      if (m > best) { best = m; bi = j; }
    }
    smax[threadIdx.x] = best; simax[threadIdx.x] = bi;
    __syncthreads();
    for (int s = blockDim.x >> 1; s > 0; s >>= 1) {
      if (threadIdx.x < s) {
        bool take = (smax[threadIdx.x + s] > smax[threadIdx.x]) ||
                    (smax[threadIdx.x + s] == smax[threadIdx.x] &&
                     simax[threadIdx.x + s] < simax[threadIdx.x]);
        if (take) { smax[threadIdx.x] = smax[threadIdx.x + s];
                    simax[threadIdx.x] = simax[threadIdx.x + s]; }
      }
      __syncthreads();
    }
    if (threadIdx.x == 0) sfar = simax[0];
    __syncthreads();
  }
}

// Gather center coordinates: dst(B,np,3) = src[b, fps_idx[b,i], :]
__global__ void k_gather3(const float* __restrict__ src, const int* __restrict__ idx,
                          float* __restrict__ dst, int Ns, int np) {
  int t = blockIdx.x * blockDim.x + threadIdx.x;
  if (t >= BATCH * np) return;
  int b = t / np;
  int j = idx[t];
  const float* s = src + (size_t)(b * Ns + j) * 3;
  dst[t * 3 + 0] = s[0]; dst[t * 3 + 1] = s[1]; dst[t * 3 + 2] = s[2];
}

// Ball query: first 32 in-radius indices (ascending), pad with first hit.
__global__ void k_ball(const float* __restrict__ xyz_src, const float* __restrict__ centers,
                       int* __restrict__ out, int Ns, int np, float r2) {
  int t = blockIdx.x * blockDim.x + threadIdx.x;
  if (t >= BATCH * np) return;
  int b = t / np;
  const float* X = xyz_src + (size_t)b * Ns * 3;
  float cx = centers[t * 3 + 0], cy = centers[t * 3 + 1], cz = centers[t * 3 + 2];
  int* o = out + (size_t)t * 32;
  int cnt = 0, first = 0;
  for (int j = 0; j < Ns && cnt < 32; ++j) {
    float dx = X[j * 3 + 0] - cx, dy = X[j * 3 + 1] - cy, dz = X[j * 3 + 2] - cz;
    if (dx * dx + dy * dy + dz * dz < r2) {
      if (cnt == 0) first = j;
      o[cnt++] = j;
    }
  }
  for (int s = cnt; s < 32; ++s) o[s] = first;
}

// Build grouped SA input H(B*np*32, Kpad) f16: [xyz - center, feats, 0-pad]
__global__ void k_group(const float* __restrict__ xyz_src, const float* __restrict__ feat_src,
                        const float* __restrict__ new_xyz, const int* __restrict__ ball,
                        f16* __restrict__ H, int Ns, int np, int Cf, int Kpad) {
  int r = blockIdx.x;                 // (b*np + i)*32 + s
  int s = r & 31, ci = r >> 5;
  int b = ci / np;
  int j = ball[(size_t)ci * 32 + s];
  const float* P = xyz_src + (size_t)(b * Ns + j) * 3;
  const float* C = new_xyz + (size_t)ci * 3;
  const float* F = feat_src + (size_t)(b * Ns + j) * Cf;
  f16* h = H + (size_t)r * Kpad;
  for (int c = threadIdx.x; c < Kpad; c += blockDim.x) {
    float v;
    if (c < 3)            v = P[c] - C[c];
    else if (c < 3 + Cf)  v = F[c - 3];
    else                  v = 0.f;
    h[c] = (f16)v;
  }
}

// ---------------------------------------------------------------------------
// Weight prep: pack W(Cout,Cin) into WMMA B-fragment order, fold BN.
// Packed layout: (((tn * (Kpad/32) + kc) * 32 + lane) * 16 + i)
//   n = tn*16 + (lane&15), k = kc*32 + (lane>>4)*16 + i
// so each lane's 16 B-operand halfs are one contiguous 32-byte chunk.
// ---------------------------------------------------------------------------
__global__ void k_prep_w(const float* __restrict__ W, const float* __restrict__ g,
                         const float* __restrict__ bb, f16* __restrict__ Bp,
                         float* __restrict__ scale, float* __restrict__ bias,
                         int Cin, int Cout, int Kpad) {
  int t = blockIdx.x * blockDim.x + threadIdx.x;
  int total = Kpad * Cout;
  if (t < total) {
    int i    = t & 15;
    int lane = (t >> 4) & 31;
    int rest = t >> 9;            // tn * nkc + kc
    int nkc  = Kpad >> 5;
    int tn   = rest / nkc, kc = rest - tn * nkc;
    int n = tn * 16 + (lane & 15);
    int k = kc * 32 + ((lane >> 4) << 4) + i;
    Bp[t] = (k < Cin) ? (f16)W[n * Cin + k] : (f16)0.f;
  }
  if (t < Cout) {
    scale[t] = g[t] * rsqrtf(1.f + 1e-5f);
    bias[t]  = bb[t];
  }
}

// ---------------------------------------------------------------------------
// WMMA GEMM: out = relu(A(M,Kpad) * W * scale + bias)
// One wave per 16x16 output tile. A chunks (16 rows x 32 halfs = 1KB) are
// double-buffered into LDS via global_load_async_to_lds_b128 (ASYNCcnt),
// overlapping the next chunk's fetch with the current WMMA. B operands come
// from the fragment-packed weight buffer as contiguous 32B vector loads.
// Fragment k order: contiguous (k = 16*half + element) in both A and B.
// ---------------------------------------------------------------------------
__global__ void k_gemm(const f16* __restrict__ A, const f16* __restrict__ Bp,
                       const float* __restrict__ scale, const float* __restrict__ bias,
                       f16* __restrict__ outH, float* __restrict__ outF,
                       int M, int N, int Kpad) {
  __shared__ f16 sA[4][2][512];        // [wave][buf][16 rows x 32 halfs]
  int wave = threadIdx.x >> 5;
  int lane = threadIdx.x & 31;
  int ntn  = N >> 4;
  int nkc  = Kpad >> 5;
  int tile = blockIdx.x * (blockDim.x >> 5) + wave;
  int total = (M >> 4) * ntn;
  if (tile >= total) return;
  int tm = tile / ntn, tn = tile - tm * ntn;
  int row  = lane & 15;
  int half = lane >> 4;                 // 0: K 0-15 piece, 1: K 16-31 piece
  int col  = tn * 16 + row;
  int aoff = half * 16;                 // A frag: lanes 16-31 hold K+16 .. K+31

  // per-lane LDS landing slot: row-major (row*32 + half*16) halfs
  unsigned dsBase[2];
  dsBase[0] = (unsigned)(uintptr_t)(&sA[wave][0][0] + (row * 32 + half * 16));
  dsBase[1] = (unsigned)(uintptr_t)(&sA[wave][1][0] + (row * 32 + half * 16));
  const f16* gBase = A + (size_t)(tm * 16 + row) * Kpad + half * 16;

  // kick chunk 0
  {
    unsigned long long ga = (unsigned long long)(uintptr_t)gBase;
    asm volatile("global_load_async_to_lds_b128 %0, %1, off\n\t"
                 "global_load_async_to_lds_b128 %0, %1, off offset:16"
                 :: "v"(dsBase[0]), "v"(ga) : "memory");
  }

  v8f acc = {0.f, 0.f, 0.f, 0.f, 0.f, 0.f, 0.f, 0.f};
  for (int kc = 0; kc < nkc; ++kc) {
    if (kc + 1 < nkc) {  // kick next chunk into the other buffer
      unsigned long long ga =
          (unsigned long long)(uintptr_t)(gBase + (size_t)(kc + 1) * 32);
      asm volatile("global_load_async_to_lds_b128 %0, %1, off\n\t"
                   "global_load_async_to_lds_b128 %0, %1, off offset:16"
                   :: "v"(dsBase[(kc + 1) & 1]), "v"(ga) : "memory");
      asm volatile("s_wait_asynccnt 2" ::: "memory");
    } else {
      asm volatile("s_wait_asynccnt 0" ::: "memory");
    }
    const f16* sa = &sA[wave][kc & 1][0] + row * 32;
    v8h lo = *(const v8h*)(sa + aoff);         // K = base+aoff .. +7
    v8h hi = *(const v8h*)(sa + aoff + 8);     // K = base+aoff+8 .. +15
    v16h afrag = __builtin_shufflevector(lo, hi, 0, 1, 2, 3, 4, 5, 6, 7,
                                         8, 9, 10, 11, 12, 13, 14, 15);
    const v16h* bp = (const v16h*)(Bp + ((((size_t)tn * nkc + kc) * 32 + lane) << 4));
    v16h bfrag = *bp;
    acc = __builtin_amdgcn_wmma_f32_16x16x32_f16(
        false, afrag, false, bfrag, (short)0, acc, false, false);
  }

  float s = scale[col], bb = bias[col];
  int rbase = tm * 16 + half * 8;
  float y[8];
#pragma unroll
  for (int v = 0; v < 8; ++v) {
    float t2 = acc[v] * s + bb;
    y[v] = t2 > 0.f ? t2 : 0.f;
    outH[(size_t)(rbase + v) * N + col] = (f16)y[v];
  }
  if (outF != nullptr) {
#pragma unroll
    for (int v = 0; v < 8; ++v)
      outF[(size_t)(rbase + v) * N + col] = y[v];
  }
}

// Max over the 32 samples of each group -> f32 level features (B,np,C)
__global__ void k_maxpool(const f16* __restrict__ H, float* __restrict__ lf,
                          int np, int C) {
  int t = blockIdx.x * blockDim.x + threadIdx.x;
  if (t >= BATCH * np * C) return;
  int c = t % C, ci = t / C;
  const f16* p = H + (size_t)ci * 32 * C + c;
  float m = -3e38f;
  for (int s = 0; s < 32; ++s) m = fmaxf(m, (float)p[(size_t)s * C]);
  lf[t] = m;
}

// ---------------------------------------------------------------------------
// FP stage: 3-NN (idx + normalized inverse-distance weights)
// ---------------------------------------------------------------------------
__global__ void k_three_nn(const float* __restrict__ ux, const float* __restrict__ kx,
                           int Nu, int Nk, int* __restrict__ idx3, float* __restrict__ w3) {
  int t = blockIdx.x * blockDim.x + threadIdx.x;
  if (t >= BATCH * Nu) return;
  int b = t / Nu;
  const float* U = ux + (size_t)t * 3;
  const float* K = kx + (size_t)b * Nk * 3;
  float ux0 = U[0], uy0 = U[1], uz0 = U[2];
  float d0 = 3e38f, d1 = 3e38f, d2 = 3e38f;
  int i0 = 0, i1 = 0, i2 = 0;
  for (int j = 0; j < Nk; ++j) {
    float dx = K[j * 3 + 0] - ux0, dy = K[j * 3 + 1] - uy0, dz = K[j * 3 + 2] - uz0;
    float d = dx * dx + dy * dy + dz * dz;
    if (d < d0)      { d2 = d1; i2 = i1; d1 = d0; i1 = i0; d0 = d; i0 = j; }
    else if (d < d1) { d2 = d1; i2 = i1; d1 = d;  i1 = j; }
    else if (d < d2) { d2 = d;  i2 = j; }
  }
  float w0 = 1.f / (d0 + 1e-8f), w1 = 1.f / (d1 + 1e-8f), w2 = 1.f / (d2 + 1e-8f);
  float ws = w0 + w1 + w2;
  idx3[t * 3 + 0] = i0; idx3[t * 3 + 1] = i1; idx3[t * 3 + 2] = i2;
  w3[t * 3 + 0] = w0 / ws; w3[t * 3 + 1] = w1 / ws; w3[t * 3 + 2] = w2 / ws;
}

// Build FP MLP input: H(B*Nu, Kpad) f16 = [interp(kf) (Ck), uf (Cu), 0-pad]
__global__ void k_fp_build(const float* __restrict__ kf, const float* __restrict__ uf,
                           const int* __restrict__ idx3, const float* __restrict__ w3,
                           f16* __restrict__ H, int Nu, int Nk, int Ck, int Cu, int Kpad) {
  int t = blockIdx.x;                 // b*Nu + u
  int b = t / Nu;
  const int*   I = idx3 + (size_t)t * 3;
  const float* W = w3  + (size_t)t * 3;
  const float* KF = kf + (size_t)b * Nk * Ck;
  const float* UF = uf + (size_t)t * Cu;
  f16* h = H + (size_t)t * Kpad;
  for (int c = threadIdx.x; c < Kpad; c += blockDim.x) {
    float v;
    if (c < Ck)
      v = W[0] * KF[(size_t)I[0] * Ck + c] + W[1] * KF[(size_t)I[1] * Ck + c] +
          W[2] * KF[(size_t)I[2] * Ck + c];
    else if (c < Ck + Cu)
      v = UF[c - Ck];
    else
      v = 0.f;
    h[c] = (f16)v;
  }
}

// ---------------------------------------------------------------------------
// Head: only point 0 of level 0 is consumed by the network, so FP0 + FC +
// critic heads are computed per-batch for a single point (8 rows total).
// ---------------------------------------------------------------------------
__device__ __forceinline__ void d_dense(const float* in, float* out,
    const float* W, const float* g, const float* bb, int cin, int cout,
    int t, int mode) {
  if (t < cout) {
    float acc = 0.f;
    for (int k = 0; k < cin; ++k) acc += in[k] * W[t * cin + k];
    float y;
    if (mode == 0) {            // BN + relu
      y = acc * (g[t] * rsqrtf(1.f + 1e-5f)) + bb[t];
      y = y > 0.f ? y : 0.f;
    } else {                    // bias + leaky relu(0.01)
      y = acc + bb[t];
      y = y > 0.f ? y : 0.01f * y;
    }
    out[t] = y;
  }
}

__global__ void k_head(const float* __restrict__ cp, const float* __restrict__ dirs,
                       const float* __restrict__ xyz1, const float* __restrict__ lf1,
                       const float* w1, const float* g1, const float* b1,
                       const float* w2, const float* g2, const float* b2,
                       const float* w3, const float* g3, const float* b3,
                       const float* wfc, const float* gfc, const float* bfc,
                       const float* wcp, const float* bcp,
                       const float* wdir, const float* bdir,
                       const float* wc1, const float* bc1,
                       const float* wc2, const float* bc2,
                       float* __restrict__ out) {
  __shared__ float s0[192], s1[192];
  __shared__ int   nnI[3];
  __shared__ float nnW[3];
  int b = blockIdx.x, t = threadIdx.x;
  float px = cp[b * 3 + 0], py = cp[b * 3 + 1], pz = cp[b * 3 + 2];
  if (t == 0) {    // 3-NN of point0 (== cp) among 1024 level-1 centers
    const float* X = xyz1 + (size_t)b * 1024 * 3;
    float d0 = 3e38f, d1 = 3e38f, d2 = 3e38f;
    int i0 = 0, i1 = 0, i2 = 0;
    for (int j = 0; j < 1024; ++j) {
      float dx = X[j * 3 + 0] - px, dy = X[j * 3 + 1] - py, dz = X[j * 3 + 2] - pz;
      float d = dx * dx + dy * dy + dz * dz;
      if (d < d0)      { d2 = d1; i2 = i1; d1 = d0; i1 = i0; d0 = d; i0 = j; }
      else if (d < d1) { d2 = d1; i2 = i1; d1 = d;  i1 = j; }
      else if (d < d2) { d2 = d;  i2 = j; }
    }
    float a0 = 1.f / (d0 + 1e-8f), a1 = 1.f / (d1 + 1e-8f), a2 = 1.f / (d2 + 1e-8f);
    float ws = a0 + a1 + a2;
    nnI[0] = i0; nnI[1] = i1; nnI[2] = i2;
    nnW[0] = a0 / ws; nnW[1] = a1 / ws; nnW[2] = a2 / ws;
  }
  __syncthreads();
  if (t < 128) {
    const float* F = lf1 + (size_t)b * 1024 * 128;
    s0[t] = nnW[0] * F[(size_t)nnI[0] * 128 + t] +
            nnW[1] * F[(size_t)nnI[1] * 128 + t] +
            nnW[2] * F[(size_t)nnI[2] * 128 + t];
  } else if (t < 132) {
    s0[t] = (t == 131) ? 1.f : cp[b * 3 + (t - 128)];   // feats0[b,0] = (cp,1)
  }
  __syncthreads();
  d_dense(s0, s1, w1, g1, b1, 132, 128, t, 0); __syncthreads();
  d_dense(s1, s0, w2, g2, b2, 128, 128, t, 0); __syncthreads();
  d_dense(s0, s1, w3, g3, b3, 128, 128, t, 0); __syncthreads();
  d_dense(s1, s0, wfc, gfc, bfc, 128, 128, t, 0); __syncthreads();   // fc -> net1
  if (t < 128)      s1[t] = s0[t];
  else if (t < 160) { int o = t - 128; float a = bcp[o];
                      for (int k = 0; k < 3; ++k) a += cp[b * 3 + k] * wcp[o * 3 + k];
                      s1[t] = a; }
  else              { int o = t - 160; float a = bdir[o];
                      for (int k = 0; k < 6; ++k) a += dirs[b * 6 + k] * wdir[o * 6 + k];
                      s1[t] = a; }
  __syncthreads();
  d_dense(s1, s0, wc1, nullptr, bc1, 192, 128, t, 1); __syncthreads();
  if (t == 0) {
    float z = bc2[0];
    for (int k = 0; k < 128; ++k) z += s0[k] * wc2[k];
    out[b] = 1.f / (1.f + expf(-z));
  }
}

// ---------------------------------------------------------------------------
// Host side
// ---------------------------------------------------------------------------
extern "C" void kernel_launch(void* const* d_in, const int* in_sizes, int n_in,
                              void* d_out, int out_size, void* d_ws, size_t ws_size,
                              hipStream_t stream) {
  (void)in_sizes; (void)n_in; (void)out_size; (void)ws_size;
  auto in = [&](int i) { return (const float*)d_in[i]; };
  const float* pcs  = in(0);
  const float* cp   = in(1);
  const float* dirs = in(2);

  char* base = (char*)d_ws;
  size_t off = 0;
  auto alloc = [&](size_t bytes) -> void* {
    void* p = base + off;
    off = (off + bytes + 255) & ~(size_t)255;
    return p;
  };

  float* xyz0  = (float*)alloc((size_t)BATCH * NPTS * 3 * 4);
  float* feat0 = (float*)alloc((size_t)BATCH * NPTS * 4 * 4);
  int*   fps1  = (int*)alloc(BATCH * 1024 * 4);
  int*   fps2  = (int*)alloc(BATCH * 256 * 4);
  int*   fps3  = (int*)alloc(BATCH * 64 * 4);
  int*   fps4  = (int*)alloc(BATCH * 16 * 4);
  float* xyz1  = (float*)alloc(BATCH * 1024 * 3 * 4);
  float* xyz2  = (float*)alloc(BATCH * 256 * 3 * 4);
  float* xyz3  = (float*)alloc(BATCH * 64 * 3 * 4);
  float* xyz4  = (float*)alloc(BATCH * 16 * 3 * 4);
  int*   ball1 = (int*)alloc((size_t)BATCH * 1024 * 32 * 4);
  int*   ball2 = (int*)alloc((size_t)BATCH * 256 * 32 * 4);
  int*   ball3 = (int*)alloc((size_t)BATCH * 64 * 32 * 4);
  int*   ball4 = (int*)alloc((size_t)BATCH * 16 * 32 * 4);
  float* lf1s  = (float*)alloc((size_t)BATCH * 1024 * 64 * 4);
  float* lf2s  = (float*)alloc((size_t)BATCH * 256 * 128 * 4);
  float* lf3s  = (float*)alloc((size_t)BATCH * 64 * 256 * 4);
  float* lf4   = (float*)alloc((size_t)BATCH * 16 * 512 * 4);
  float* lf3f  = (float*)alloc((size_t)BATCH * 64 * 256 * 4);
  float* lf2f  = (float*)alloc((size_t)BATCH * 256 * 256 * 4);
  float* lf1f  = (float*)alloc((size_t)BATCH * 1024 * 128 * 4);
  int*   nni   = (int*)alloc((size_t)BATCH * 1024 * 3 * 4);
  float* nnw   = (float*)alloc((size_t)BATCH * 1024 * 3 * 4);

  // 18 WMMA layers: param-index, Cin, Cout, Kpad(Cin rounded up to 32)
  struct LD { int wi, cin, cout, kpad; };
  static const LD layers[18] = {
      {3, 7, 32, 32},    {6, 32, 32, 32},    {9, 32, 64, 32},      // SA1
      {12, 67, 64, 96},  {15, 64, 64, 64},   {18, 64, 128, 64},    // SA2
      {21, 131, 128, 160},{24, 128, 128, 128},{27, 128, 256, 128}, // SA3
      {30, 259, 256, 288},{33, 256, 256, 256},{36, 256, 512, 256}, // SA4
      {60, 768, 256, 768},{63, 256, 256, 256},                     // FP3
      {54, 384, 256, 384},{57, 256, 256, 256},                     // FP2
      {48, 320, 256, 320},{51, 256, 128, 256},                     // FP1
  };
  f16*   Wb[18]; float* Sc[18]; float* Bi[18];
  for (int l = 0; l < 18; ++l) {
    Wb[l] = (f16*)alloc((size_t)layers[l].kpad * layers[l].cout * 2);
    Sc[l] = (float*)alloc(layers[l].cout * 4);
    Bi[l] = (float*)alloc(layers[l].cout * 4);
  }
  f16* actA = (f16*)alloc((size_t)20 * 1024 * 1024);
  f16* actB = (f16*)alloc((size_t)36 * 1024 * 1024);

  for (int l = 0; l < 18; ++l) {
    int total = layers[l].kpad * layers[l].cout;
    k_prep_w<<<(total + 255) / 256, 256, 0, stream>>>(
        in(layers[l].wi), in(layers[l].wi + 1), in(layers[l].wi + 2),
        Wb[l], Sc[l], Bi[l], layers[l].cin, layers[l].cout, layers[l].kpad);
  }

  auto gemm = [&](const f16* A, int l, f16* oH, float* oF, int M, int Nc, int Kp) {
    int tiles = (M / 16) * (Nc / 16);
    k_gemm<<<(tiles + 3) / 4, 128, 0, stream>>>(A, Wb[l], Sc[l], Bi[l], oH, oF, M, Nc, Kp);
  };

  k_pre<<<(BATCH * NPTS + 255) / 256, 256, 0, stream>>>(pcs, cp, xyz0, feat0);

  // ---- SA1: 4096 -> 1024 centers, [7->32->32->64] ----
  k_fps<<<BATCH, 256, 0, stream>>>(xyz0, fps1, 4096, 1024);
  k_gather3<<<(BATCH * 1024 + 255) / 256, 256, 0, stream>>>(xyz0, fps1, xyz1, 4096, 1024);
  k_ball<<<(BATCH * 1024 + 255) / 256, 256, 0, stream>>>(xyz0, xyz1, ball1, 4096, 1024, 0.01f);
  k_group<<<BATCH * 1024 * 32, 64, 0, stream>>>(xyz0, feat0, xyz1, ball1, actA, 4096, 1024, 4, 32);
  gemm(actA, 0, actB, nullptr, 262144, 32, 32);
  gemm(actB, 1, actA, nullptr, 262144, 32, 32);
  gemm(actA, 2, actB, nullptr, 262144, 64, 32);
  k_maxpool<<<(BATCH * 1024 * 64 + 255) / 256, 256, 0, stream>>>(actB, lf1s, 1024, 64);

  // ---- SA2: 1024 -> 256, [67->64->64->128] ----
  k_fps<<<BATCH, 256, 0, stream>>>(xyz1, fps2, 1024, 256);
  k_gather3<<<(BATCH * 256 + 255) / 256, 256, 0, stream>>>(xyz1, fps2, xyz2, 1024, 256);
  k_ball<<<(BATCH * 256 + 255) / 256, 256, 0, stream>>>(xyz1, xyz2, ball2, 1024, 256, 0.04f);
  k_group<<<BATCH * 256 * 32, 64, 0, stream>>>(xyz1, lf1s, xyz2, ball2, actA, 1024, 256, 64, 96);
  gemm(actA, 3, actB, nullptr, 65536, 64, 96);
  gemm(actB, 4, actA, nullptr, 65536, 64, 64);
  gemm(actA, 5, actB, nullptr, 65536, 128, 64);
  k_maxpool<<<(BATCH * 256 * 128 + 255) / 256, 256, 0, stream>>>(actB, lf2s, 256, 128);

  // ---- SA3: 256 -> 64, [131->128->128->256] ----
  k_fps<<<BATCH, 256, 0, stream>>>(xyz2, fps3, 256, 64);
  k_gather3<<<(BATCH * 64 + 255) / 256, 256, 0, stream>>>(xyz2, fps3, xyz3, 256, 64);
  k_ball<<<(BATCH * 64 + 255) / 256, 256, 0, stream>>>(xyz2, xyz3, ball3, 256, 64, 0.16f);
  k_group<<<BATCH * 64 * 32, 64, 0, stream>>>(xyz2, lf2s, xyz3, ball3, actA, 256, 64, 128, 160);
  gemm(actA, 6, actB, nullptr, 16384, 128, 160);
  gemm(actB, 7, actA, nullptr, 16384, 128, 128);
  gemm(actA, 8, actB, nullptr, 16384, 256, 128);
  k_maxpool<<<(BATCH * 64 * 256 + 255) / 256, 256, 0, stream>>>(actB, lf3s, 64, 256);

  // ---- SA4: 64 -> 16, [259->256->256->512] ----
  k_fps<<<BATCH, 256, 0, stream>>>(xyz3, fps4, 64, 16);
  k_gather3<<<(BATCH * 16 + 255) / 256, 256, 0, stream>>>(xyz3, fps4, xyz4, 64, 16);
  k_ball<<<(BATCH * 16 + 255) / 256, 256, 0, stream>>>(xyz3, xyz4, ball4, 64, 16, 0.64f);
  k_group<<<BATCH * 16 * 32, 64, 0, stream>>>(xyz3, lf3s, xyz4, ball4, actA, 64, 16, 256, 288);
  gemm(actA, 9, actB, nullptr, 4096, 256, 288);
  gemm(actB, 10, actA, nullptr, 4096, 256, 256);
  gemm(actA, 11, actB, nullptr, 4096, 512, 256);
  k_maxpool<<<(BATCH * 16 * 512 + 255) / 256, 256, 0, stream>>>(actB, lf4, 16, 512);

  // ---- FP3: known=16 -> unknown=64, [768->256->256] ----
  k_three_nn<<<(BATCH * 64 + 255) / 256, 256, 0, stream>>>(xyz3, xyz4, 64, 16, nni, nnw);
  k_fp_build<<<BATCH * 64, 128, 0, stream>>>(lf4, lf3s, nni, nnw, actA, 64, 16, 512, 256, 768);
  gemm(actA, 12, actB, nullptr, 512, 256, 768);
  gemm(actB, 13, actA, lf3f, 512, 256, 256);

  // ---- FP2: 64 -> 256, [384->256->256] ----
  k_three_nn<<<(BATCH * 256 + 255) / 256, 256, 0, stream>>>(xyz2, xyz3, 256, 64, nni, nnw);
  k_fp_build<<<BATCH * 256, 128, 0, stream>>>(lf3f, lf2s, nni, nnw, actA, 256, 64, 256, 128, 384);
  gemm(actA, 14, actB, nullptr, 2048, 256, 384);
  gemm(actB, 15, actA, lf2f, 2048, 256, 256);

  // ---- FP1: 256 -> 1024, [320->256->128] ----
  k_three_nn<<<(BATCH * 1024 + 255) / 256, 256, 0, stream>>>(xyz1, xyz2, 1024, 256, nni, nnw);
  k_fp_build<<<BATCH * 1024, 128, 0, stream>>>(lf2f, lf1s, nni, nnw, actA, 1024, 256, 256, 64, 320);
  gemm(actA, 16, actB, nullptr, 8192, 256, 320);
  gemm(actB, 17, actA, lf1f, 8192, 128, 256);

  // ---- Head: FP0 at point 0 only + fc + critic (per batch) ----
  k_head<<<BATCH, 192, 0, stream>>>(
      cp, dirs, xyz1, lf1f,
      in(39), in(40), in(41),      // fp0 layer1 132->128
      in(42), in(43), in(44),      // fp0 layer2 128->128
      in(45), in(46), in(47),      // fp0 layer3 128->128
      in(66), in(67), in(68),      // fc 128->128
      in(69), in(70),              // mlp_cp
      in(71), in(72),              // mlp_dir
      in(73), in(74),              // critic1
      in(75), in(76),              // critic2
      (float*)d_out);
}
